// AIGTransformer_60000693125877
// MI455X (gfx1250) — matrix operand order repaired and checked by reading, not compiled, not run on hardware
//
#include <hip/hip_runtime.h>

// ---------------- problem constants ----------------
#define G_    2048
#define N_    120
#define D_    64
#define H_    2
#define DH_   32
#define EPER_ 240
#define NF_   4
#define EF_   2

// ---------------- LDS layout (bytes; every region 16B aligned) ----------------
// f32 strides padded (68/132) to dodge bank conflicts; bf16 strides padded to a
// multiple of 8 elements (72/136/264) so fragment runs stay 16B-aligned.
#define LDH    68      // f32 stride, 64-wide (h, t)
#define LDSC   132     // f32 stride, 128-wide (scores)
#define LHN    72      // bf16 stride, 64-wide (hn, q, k, W*t[.][64])
#define LVT    136     // bf16 stride, 128-wide (vT, pb)
#define LFF    264     // bf16 stride, 256-wide (ff, W2t rows)

#define OFF_H    0                         // h  : 128*68*4  = 34816
#define OFF_HN   34816                     // hn : 128*72*2  = 18432
#define OFF_W    53248                     // weights region : 70656
                                           //   phase QKV : Wq_t,Wk_t,Wv_t  3*[64][72]*2 = 27648
                                           //   phase FF  : W1t [256][72]*2=36864 + W2t [64][264]*2=33792
                                           //   phase head: np1t [64][72]*2 = 9216
#define OFF_U    123904                    // union region : 121856
#define OFF_Q    (OFF_U)                   // q  : 128*72*2 = 18432
#define OFF_K    (OFF_U + 18432)           // k  : 128*72*2 = 18432
#define OFF_VT   (OFF_U + 36864)           // vT : 64*136*2 = 17408
#define OFF_S    (OFF_U + 54272)           // sc : 128*132*4 = 67584
// ff (128*264*2=67584) and t (128*68*4=34816) overlay OFF_U after attention.
#define OFF_PB   245760                    // pb : 128*136*2 = 34816 (bf16 softmax probs)
#define SMEM_BYTES 280576                  // 274 KB  (< 320 KB / WGP)

typedef __attribute__((ext_vector_type(16))) __bf16 v16bf;
typedef __attribute__((ext_vector_type(8)))  float  v8f;

union BF16x16 { v16bf v; uint4 q[2]; };

__device__ __forceinline__ __bf16 f2bf(float f) {
  unsigned u = __builtin_bit_cast(unsigned, f);
  unsigned short hs = (unsigned short)(u >> 16);
  return __builtin_bit_cast(__bf16, hs);
}

// ---------------- generic tiled WMMA GEMM ----------------
// A element (m,kk): aMode 0 -> f32 A32[m*lda + k0base + kk] (convert on load)
//                   aMode 1 -> bf16 A16[m*lda + k0base + kk] (fragment = 2x ds_load_b128)
// B element (kk,n): bf16 B[(brow_off+n)*bstride + bk0 + kk]  (fragment = 2x ds_load_b128)
// store sMode: 0 f32 store, 1 f32 store+relu, 2 f32 accumulate,
//              3 bf16 store, 4 bf16 store+relu, 5 bf16 transposed store (C16[n*ldc+m])
// Mode args are compile-time literals at call sites -> dead code folds away.
__device__ __forceinline__ void gemm_tiles(
    int aMode, const float* __restrict__ A32, const __bf16* __restrict__ A16,
    int lda, int k0base,
    const __bf16* __restrict__ B, int bstride, int brow_off, int bk0,
    int sMode, float* __restrict__ C32, __bf16* __restrict__ C16, int ldc, int col_off,
    int Mtiles, int Ntiles, int Ksteps,
    const float* __restrict__ bias, float scale,
    int wave, int lane)
{
  const int mlane = lane & 15;
  const int hi    = lane >> 4;
  const int total = Mtiles * Ntiles;
  #pragma unroll 2                     // 2 independent accumulators: loads overlap WMMA
  for (int tIdx = wave; tIdx < total; tIdx += 8) {
    const int mt = tIdx / Ntiles;
    const int nt = tIdx - mt * Ntiles;
    const int row0 = mt * 16, col0 = nt * 16;
    v8f acc = {};
    for (int ks = 0; ks < Ksteps; ++ks) {
      const int k0 = k0base + ks * 32;
      v16bf a;
      if (aMode == 0) {           // f32 source, convert (h only)
        const float* ar = A32 + (row0 + mlane) * lda + k0;
        #pragma unroll
        for (int e = 0; e < 8; ++e) {
          const int kb = ((e >> 2) * 16) + hi * 8 + (e & 3) * 2;
          a[2 * e]     = f2bf(ar[kb]);
          a[2 * e + 1] = f2bf(ar[kb + 1]);
        }
      } else {                    // packed bf16 source: two contiguous 16B runs
        const __bf16* ar = A16 + (row0 + mlane) * lda + k0;
        BF16x16 t;
        t.q[0] = *(const uint4*)(ar + hi * 8);
        t.q[1] = *(const uint4*)(ar + 16 + hi * 8);
        a = t.v;
      }
      // B fragment: lane holds col (col0+mlane), K runs [hi*8,+8) and [16+hi*8,+8)
      const __bf16* bp = B + (brow_off + col0 + mlane) * bstride + bk0 + ks * 32;
      BF16x16 tb;
      tb.q[0] = *(const uint4*)(bp + hi * 8);
      tb.q[1] = *(const uint4*)(bp + 16 + hi * 8);
      acc = __builtin_amdgcn_wmma_f32_16x16x32_bf16(
          false, a, false, tb.v, (short)0, acc, false, false);
    }
    const int ncol  = col0 + mlane;
    const int rbase = row0 + hi * 8;
    const float bv = bias ? bias[ncol] : 0.0f;
    #pragma unroll
    for (int r = 0; r < 8; ++r) {
      float val = acc[r] * scale + bv;
      if (sMode == 1 || sMode == 4) val = fmaxf(val, 0.0f);
      if (sMode <= 1)      C32[(rbase + r) * ldc + col_off + ncol] = val;
      else if (sMode == 2) C32[(rbase + r) * ldc + col_off + ncol] += val;
      else if (sMode <= 4) C16[(rbase + r) * ldc + ncol] = f2bf(val);
      else                 C16[ncol * ldc + rbase + r] = f2bf(val);   // vT
    }
  }
}

// ---------------- weight staging: global f32 [K][Nout] -> LDS bf16 [Nout][stride] ----------------
__device__ __forceinline__ void stage_wt(const float* __restrict__ W, int K, int noutShift,
                                         __bf16* __restrict__ dst, int stride, int tid) {
  const int total = K << noutShift;
  const int mask = (1 << noutShift) - 1;
  for (int i = tid; i < total; i += 256) {       // coalesced global reads, one pass per block
    const int kk = i >> noutShift;
    const int n  = i & mask;
    dst[n * stride + kk] = f2bf(W[i]);
  }
}

// ---------------- LayerNorm: wave-per-row, writes packed bf16 ----------------
__device__ __forceinline__ void layer_norm(
    const float* __restrict__ src, __bf16* __restrict__ dst,
    const float* __restrict__ gam, const float* __restrict__ bet,
    int wave, int lane)
{
  for (int r = wave; r < N_; r += 8) {
    float x0 = src[r * LDH + lane];
    float x1 = src[r * LDH + lane + 32];
    float s = x0 + x1;
    #pragma unroll
    for (int o = 16; o > 0; o >>= 1) s += __shfl_xor(s, o, 32);
    const float mean = s * (1.0f / 64.0f);
    const float d0 = x0 - mean, d1 = x1 - mean;
    float vv = d0 * d0 + d1 * d1;
    #pragma unroll
    for (int o = 16; o > 0; o >>= 1) vv += __shfl_xor(vv, o, 32);
    const float rstd = rsqrtf(vv * (1.0f / 64.0f) + 1e-5f);
    dst[r * LHN + lane]      = f2bf(d0 * rstd * gam[lane]      + bet[lane]);
    dst[r * LHN + lane + 32] = f2bf(d1 * rstd * gam[lane + 32] + bet[lane + 32]);
  }
}

// ------- softmax: read f32 scores, write normalized probs as packed bf16 -------
// pad cols hold -1e30 -> exp underflows to exact 0 -> pb pad cols become bf16 zero.
__device__ __forceinline__ void softmax_rows(const float* __restrict__ s,
                                             __bf16* __restrict__ pb,
                                             int wave, int lane) {
  for (int r = wave; r < N_; r += 8) {
    float x[4];
    #pragma unroll
    for (int j = 0; j < 4; ++j) x[j] = s[r * LDSC + lane + 32 * j];
    float mx = fmaxf(fmaxf(x[0], x[1]), fmaxf(x[2], x[3]));
    #pragma unroll
    for (int o = 16; o > 0; o >>= 1) mx = fmaxf(mx, __shfl_xor(mx, o, 32));
    float sum = 0.0f;
    #pragma unroll
    for (int j = 0; j < 4; ++j) { x[j] = __expf(x[j] - mx); sum += x[j]; }
    #pragma unroll
    for (int o = 16; o > 0; o >>= 1) sum += __shfl_xor(sum, o, 32);
    const float inv = 1.0f / sum;
    #pragma unroll
    for (int j = 0; j < 4; ++j) pb[r * LVT + lane + 32 * j] = f2bf(x[j] * inv);
  }
}

// ---------------- parameter bundle ----------------
struct LayerP {
  const float *W1, *W2, *We, *Wk, *Wq, *Wv;
  const float *b1, *b2, *be, *bk, *bq, *bv;
  const float *ln1_b, *ln1_g, *ln2_b, *ln2_g;
};
struct KParams {
  const float* x;  const int* ei;  const float* ea;
  const float *emb_W, *emb_b;
  LayerP L[2];
  const float *np1_W, *np1_b, *np2_W, *np2_b, *pos;
  float* out;
};

__global__ __launch_bounds__(256, 1) void aig_transformer_kernel(KParams p) {
  extern __shared__ char smraw[];
  float*   h     = (float*)  (smraw + OFF_H);
  __bf16*  hn    = (__bf16*) (smraw + OFF_HN);
  __bf16*  wA    = (__bf16*) (smraw + OFF_W);              // Wq_t | W1t | np1t
  __bf16*  wB    = wA + 64 * LHN;                          // Wk_t
  __bf16*  wC    = wB + 64 * LHN;                          // Wv_t
  __bf16*  w2t   = (__bf16*) (smraw + OFF_W + 36864);      // W2t (FF phase)
  __bf16*  qb    = (__bf16*) (smraw + OFF_Q);
  __bf16*  kb    = (__bf16*) (smraw + OFF_K);
  __bf16*  vT    = (__bf16*) (smraw + OFF_VT);
  float*   sc    = (float*)  (smraw + OFF_S);
  __bf16*  pb    = (__bf16*) (smraw + OFF_PB);             // bf16 probabilities
  __bf16*  ffb   = (__bf16*) (smraw + OFF_U);              // overlays q/k/vT/sc
  float*   t32   = (float*)  (smraw + OFF_U);              // head temp

  const int g    = blockIdx.x;
  const int tid  = threadIdx.x;
  const int wave = tid >> 5;
  const int lane = tid & 31;

  // ---- embedding + positional ----
  for (int i = tid; i < N_ * D_; i += 256) {
    const int n = i >> 6, c = i & 63;
    const float* xr = p.x + (size_t)(g * N_ + n) * NF_;
    float acc = p.emb_b[c] + p.pos[n * D_ + c];
    #pragma unroll
    for (int j = 0; j < NF_; ++j) acc += xr[j] * p.emb_W[j * D_ + c];
    h[n * LDH + c] = acc;
  }
  // zero pad rows 120..127 of h (f32), hn (bf16), pb (bf16, zeroed once: region never overlaid)
  for (int i = tid; i < 8 * LDH; i += 256)
    h[(N_ + i / LDH) * LDH + (i % LDH)] = 0.0f;
  for (int i = tid; i < 8 * LHN; i += 256)
    hn[(N_ + i / LHN) * LHN + (i % LHN)] = f2bf(0.0f);
  for (int i = tid; i < 8 * LVT; i += 256)
    pb[(N_ + i / LVT) * LVT + (i % LVT)] = f2bf(0.0f);
  __syncthreads();

  // ---- 2 transformer layers ----
  for (int li = 0; li < 2; ++li) {
    const LayerP& L = p.L[li];

    // stage QKV weights (bf16, transposed [64][72]) — once per block
    stage_wt(L.Wq, D_, 6, wA, LHN, tid);
    stage_wt(L.Wk, D_, 6, wB, LHN, tid);
    stage_wt(L.Wv, D_, 6, wC, LHN, tid);
    layer_norm(h, hn, L.ln1_g, L.ln1_b, wave, lane);
    __syncthreads();

    // q/k = hn @ W (bf16 out, packed); v stored transposed vT[feat][node]
    gemm_tiles(1, nullptr, hn, LHN, 0, wA, LHN, 0, 0, 3, nullptr, qb, LHN, 0,
               8, 4, 2, L.bq, 1.0f, wave, lane);
    gemm_tiles(1, nullptr, hn, LHN, 0, wB, LHN, 0, 0, 3, nullptr, kb, LHN, 0,
               8, 4, 2, L.bk, 1.0f, wave, lane);
    gemm_tiles(1, nullptr, hn, LHN, 0, wC, LHN, 0, 0, 5, nullptr, vT, LVT, 0,
               8, 4, 2, L.bv, 1.0f, wave, lane);
    __syncthreads();

    // stage FF weights while QKV region is free (consumed by FF after attention)
    stage_wt(L.W1, D_, 8, wA, LHN, tid);          // W1t: [256][72]
    stage_wt(L.W2, 4 * D_, 6, w2t, LFF, tid);     // W2t: [64][264]

    for (int hd = 0; hd < H_; ++hd) {
      // scores = (q_h @ k_h^T)/sqrt(32)  — B = k rows (contiguous K run per lane)
      gemm_tiles(1, nullptr, qb, LHN, hd * DH_, kb, LHN, 0, hd * DH_,
                 0, sc, nullptr, LDSC, 0, 8, 8, 1, nullptr, 0.17677669529663687f, wave, lane);
      __syncthreads();

      // mask padded key columns so softmax weight is exactly 0
      for (int i = tid; i < 128 * 8; i += 256)
        sc[(i >> 3) * LDSC + N_ + (i & 7)] = -1e30f;
      __syncthreads();

      // edge-bias scatter-add (dups -> ds_add_f32)
      for (int e = tid; e < EPER_; e += 256) {
        const int base = g * EPER_ + e;
        const int es = p.ei[base] % N_;
        const int ed = p.ei[(size_t)G_ * EPER_ + base] % N_;
        const float a0 = p.ea[(size_t)base * EF_];
        const float a1 = p.ea[(size_t)base * EF_ + 1];
        const float w = a0 * L.We[0 * H_ + hd] + a1 * L.We[1 * H_ + hd] + L.be[hd];
        atomicAdd(&sc[es * LDSC + ed], w);
      }
      __syncthreads();

      softmax_rows(sc, pb, wave, lane);           // f32 scores -> packed bf16 probs
      __syncthreads();

      // h[:, hd*32:+32] += probs @ v_h  (A from pb, B from vT: all-b128 fast path)
      gemm_tiles(1, nullptr, pb, LVT, 0, vT, LVT, hd * DH_, 0,
                 2, h, nullptr, LDH, hd * DH_, 8, 2, 4, nullptr, 1.0f, wave, lane);
      __syncthreads();
    }

    layer_norm(h, hn, L.ln2_g, L.ln2_b, wave, lane);
    __syncthreads();

    // ff = relu(hn @ W1 + b1)  (bf16 out, overlays q/k/vT/sc)
    gemm_tiles(1, nullptr, hn, LHN, 0, wA, LHN, 0, 0, 4, nullptr, ffb, LFF, 0,
               8, 16, 2, L.b1, 1.0f, wave, lane);
    __syncthreads();

    // h += ff @ W2 + b2
    gemm_tiles(1, nullptr, ffb, LFF, 0, w2t, LFF, 0, 0, 2, h, nullptr, LDH, 0,
               8, 4, 8, L.b2, 1.0f, wave, lane);
    __syncthreads();
  }

  // ---- node head ----
  stage_wt(p.np1_W, D_, 6, wA, LHN, tid);
  __syncthreads();
  gemm_tiles(0, h, nullptr, LDH, 0, wA, LHN, 0, 0, 1, t32, nullptr, LDH, 0,
             8, 4, 2, p.np1_b, 1.0f, wave, lane);
  __syncthreads();

  // out = relu(...) @ np2_W + np2_b   (N-out = 4: scalar dot, np2_W is L2-hot)
  for (int i = tid; i < N_ * NF_; i += 256) {
    const int n = i >> 2, f = i & 3;
    float acc = p.np2_b[f];
    #pragma unroll 8
    for (int j = 0; j < D_; ++j) acc += t32[n * LDH + j] * p.np2_W[j * NF_ + f];
    p.out[(size_t)(g * N_ + n) * NF_ + f] = acc;
  }
}

extern "C" void kernel_launch(void* const* d_in, const int* in_sizes, int n_in,
                              void* d_out, int out_size, void* d_ws, size_t ws_size,
                              hipStream_t stream) {
  (void)in_sizes; (void)n_in; (void)out_size; (void)d_ws; (void)ws_size;
  // Input order: x, edge_index, edge_attr, then params flattened with sorted dict keys:
  // emb_W, emb_b, layers[i]{W1,W2,We,Wk,Wq,Wv,b1,b2,be,bk,bq,bv,ln1_b,ln1_g,ln2_b,ln2_g},
  // np1_W, np1_b, np2_W, np2_b, pos
  KParams P;
  int i = 0;
  P.x     = (const float*)d_in[i++];
  P.ei    = (const int*)  d_in[i++];
  P.ea    = (const float*)d_in[i++];
  P.emb_W = (const float*)d_in[i++];
  P.emb_b = (const float*)d_in[i++];
  for (int l = 0; l < 2; ++l) {
    LayerP& L = P.L[l];
    L.W1    = (const float*)d_in[i++];
    L.W2    = (const float*)d_in[i++];
    L.We    = (const float*)d_in[i++];
    L.Wk    = (const float*)d_in[i++];
    L.Wq    = (const float*)d_in[i++];
    L.Wv    = (const float*)d_in[i++];
    L.b1    = (const float*)d_in[i++];
    L.b2    = (const float*)d_in[i++];
    L.be    = (const float*)d_in[i++];
    L.bk    = (const float*)d_in[i++];
    L.bq    = (const float*)d_in[i++];
    L.bv    = (const float*)d_in[i++];
    L.ln1_b = (const float*)d_in[i++];
    L.ln1_g = (const float*)d_in[i++];
    L.ln2_b = (const float*)d_in[i++];
    L.ln2_g = (const float*)d_in[i++];
  }
  P.np1_W = (const float*)d_in[i++];
  P.np1_b = (const float*)d_in[i++];
  P.np2_W = (const float*)d_in[i++];
  P.np2_b = (const float*)d_in[i++];
  P.pos   = (const float*)d_in[i++];
  P.out   = (float*)d_out;

  (void)hipFuncSetAttribute((const void*)aig_transformer_kernel,
                            hipFuncAttributeMaxDynamicSharedMemorySize, SMEM_BYTES);
  aig_transformer_kernel<<<dim3(G_), dim3(256), SMEM_BYTES, stream>>>(P);
}
